// EchoStateNetwork_78340203479092
// MI455X (gfx1250) — matrix operand order repaired and checked by reading, not compiled, not run on hardware
//
#include <hip/hip_runtime.h>
#include <hip/hip_bf16.h>
#include <math.h>

// ---------------------------------------------------------------------------
// Echo State Network on MI455X (gfx1250)
//   - persistent kernel, 64 WGs (1 per WGP), grid-wide epoch barrier per step
//   - W_res held in LDS as split bf16 (hi + lo), 256KB per WGP
//   - matvec via v_wmma_f32_16x16x32_bf16, 3-term split for fp32-class accuracy
// ---------------------------------------------------------------------------

#define N_NODES      2048
#define T_STEPS      8192
#define LEAK         0.99f

#define NUM_WG        64
#define WAVES_PER_WG   2
#define THREADS_PER_WG (WAVES_PER_WG * 32)
#define ROWS_PER_WAVE 16
#define ROWS_PER_WG   (WAVES_PER_WG * ROWS_PER_WAVE)   // 32
#define TOTAL_WAVES   (NUM_WG * WAVES_PER_WG)          // 128
#define K_CHUNK       32

// LDS partition (bytes)
#define LDS_WHI_OFF   0
#define LDS_WLO_OFF   (ROWS_PER_WG * N_NODES * 2)              // 131072
#define LDS_SHI_OFF   (LDS_WLO_OFF + ROWS_PER_WG * N_NODES * 2) // 262144
#define LDS_SLO_OFF   (LDS_SHI_OFF + N_NODES * 2)               // 266240
#define LDS_BYTES     (LDS_SLO_OFF + N_NODES * 2)               // 270336

typedef __bf16 bf16_t;
typedef bf16_t v16bf __attribute__((ext_vector_type(16)));
typedef float  v8f   __attribute__((ext_vector_type(8)));
typedef unsigned int u32;
typedef u32 u32x4 __attribute__((ext_vector_type(4)));
typedef u32 u32x8 __attribute__((ext_vector_type(8)));

// Load 16 bf16 as two contiguous 16-byte runs (=> 2x ds_load_b128)
__device__ __forceinline__ v16bf load_2x8(const bf16_t* p0, const bf16_t* p1) {
    u32x4 a = *(const u32x4*)p0;
    u32x4 b = *(const u32x4*)p1;
    u32x8 r;
    r[0] = a[0]; r[1] = a[1]; r[2] = a[2]; r[3] = a[3];
    r[4] = b[0]; r[5] = b[1]; r[6] = b[2]; r[7] = b[3];
    return __builtin_bit_cast(v16bf, r);
}

__global__ void esn_init(const float* __restrict__ state0,
                         float* __restrict__ stateA, u32* __restrict__ cnt) {
    int i = blockIdx.x * blockDim.x + threadIdx.x;
    if (i < N_NODES) stateA[i] = state0[i];
    if (i == 0) *cnt = 0u;
}

__global__ void __launch_bounds__(THREADS_PER_WG)
esn_persistent(const float* __restrict__ X, const float* __restrict__ Y,
               const float* __restrict__ W_in, const float* __restrict__ W_res,
               const float* __restrict__ W_fb, const float* __restrict__ W_out,
               float* __restrict__ stateA, float* __restrict__ stateB,
               float* __restrict__ partials, u32* __restrict__ cnt) {
    extern __shared__ char lds_raw[];
    bf16_t* Whi = (bf16_t*)(lds_raw + LDS_WHI_OFF);
    bf16_t* Wlo = (bf16_t*)(lds_raw + LDS_WLO_OFF);
    bf16_t* shi = (bf16_t*)(lds_raw + LDS_SHI_OFF);
    bf16_t* slo = (bf16_t*)(lds_raw + LDS_SLO_OFF);

    const int wg   = blockIdx.x;
    const int tid  = threadIdx.x;
    const int lane = tid & 31;
    const int wid  = tid >> 5;

    // ---- one-time: load this WG's 32 rows of W_res, split into bf16 hi+lo
    const size_t row0 = (size_t)wg * ROWS_PER_WG;
    for (int i = tid; i < ROWS_PER_WG * N_NODES; i += THREADS_PER_WG) {
        float w = W_res[row0 * N_NODES + i];
        bf16_t hi = (bf16_t)w;
        Whi[i] = hi;
        Wlo[i] = (bf16_t)(w - (float)hi);
    }
    __syncthreads();

    // per-lane constants
    const int gwave = wg * WAVES_PER_WG + wid;          // 0..127
    const int col   = lane & 15;                        // output row within tile
    const int myrow = gwave * ROWS_PER_WAVE + col;      // lanes 16-31 duplicate
    const int khalf = (lane >> 4) << 4;                 // 0 or 16 (B k-half)
    const int a_off = (lane >> 4) << 3;                 // 0 or 8  (unused)
    const float wout = W_out[1 + myrow];
    const float win  = W_in[myrow];
    const float wfb  = W_fb[myrow];
    const bf16_t* bhi_base = Whi + (size_t)(wid * ROWS_PER_WAVE + col) * N_NODES + khalf;
    const bf16_t* blo_base = Wlo + (size_t)(wid * ROWS_PER_WAVE + col) * N_NODES + khalf;

    for (int t = 0; t < T_STEPS; ++t) {
        const float* scur = (t & 1) ? stateB : stateA;
        float*       snxt = (t & 1) ? stateA : stateB;

        // ---- broadcast state into LDS as split bf16
        for (int i = tid; i < N_NODES; i += THREADS_PER_WG) {
            float s = scur[i];
            bf16_t h = (bf16_t)s;
            shi[i] = h;
            slo[i] = (bf16_t)(s - (float)h);
        }
        __syncthreads();

        // ---- 16 dot products per wave via WMMA, split-bf16 3-term accumulate
        v8f c = {0.f, 0.f, 0.f, 0.f, 0.f, 0.f, 0.f, 0.f};
        for (int kb = 0; kb < N_NODES; kb += K_CHUNK) {
            v16bf a_hi = load_2x8(shi + kb + khalf, shi + kb + khalf + 8);
            v16bf a_lo = load_2x8(slo + kb + khalf, slo + kb + khalf + 8);
            v16bf b_hi = load_2x8(bhi_base + kb, bhi_base + kb + 8);
            v16bf b_lo = load_2x8(blo_base + kb, blo_base + kb + 8);
            c = __builtin_amdgcn_wmma_f32_16x16x32_bf16(false, a_hi, false, b_hi,
                                                        (short)0, c, false, false);
            c = __builtin_amdgcn_wmma_f32_16x16x32_bf16(false, a_lo, false, b_hi,
                                                        (short)0, c, false, false);
            c = __builtin_amdgcn_wmma_f32_16x16x32_bf16(false, a_hi, false, b_lo,
                                                        (short)0, c, false, false);
        }
        // D[M][N] identical over M (A rows replicated): c[0] = dot(W row `col`)
        float dot = c[0];

        float xt  = X[t];
        float yt  = Y[t];
        float old = scur[myrow];
        float pre = dot + xt * win + yt * wfb;
        float ns  = LEAK * tanhf(pre) + (1.0f - LEAK) * old;
        if (lane < 16) snxt[myrow] = ns;

        // readout partial for this wave's 16 rows
        float v = wout * ns;
        for (int off = 8; off >= 1; off >>= 1) v += __shfl_xor(v, off, 32);
        if (lane == 0) partials[(size_t)t * TOTAL_WAVES + gwave] = v;

        // ---- device-wide epoch barrier
        __threadfence();
        __syncthreads();
        if (tid == 0) {
            atomicAdd(cnt, 1u);
            const u32 target = (u32)(t + 1) * NUM_WG;
            while (__hip_atomic_load(cnt, __ATOMIC_ACQUIRE,
                                     __HIP_MEMORY_SCOPE_AGENT) < target) {
                __builtin_amdgcn_s_sleep(2);
            }
        }
        __syncthreads();
        __threadfence();
    }
}

__global__ void esn_readout(const float* __restrict__ X,
                            const float* __restrict__ W_out,
                            const float* __restrict__ b_out,
                            const float* __restrict__ partials,
                            float* __restrict__ out) {
    int t = blockIdx.x * blockDim.x + threadIdx.x;
    if (t >= T_STEPS) return;
    float acc = b_out[0] + X[t] * W_out[0];
    const float* p = partials + (size_t)t * TOTAL_WAVES;
    #pragma unroll 8
    for (int w = 0; w < TOTAL_WAVES; ++w) acc += p[w];
    out[t] = acc;
}

extern "C" void kernel_launch(void* const* d_in, const int* in_sizes, int n_in,
                              void* d_out, int out_size, void* d_ws, size_t ws_size,
                              hipStream_t stream) {
    const float* X      = (const float*)d_in[0];
    const float* Y      = (const float*)d_in[1];
    const float* W_in   = (const float*)d_in[2];
    const float* W_res  = (const float*)d_in[3];
    const float* W_fb   = (const float*)d_in[4];
    const float* W_out  = (const float*)d_in[5];
    const float* b_out  = (const float*)d_in[6];
    const float* state0 = (const float*)d_in[7];
    float* out = (float*)d_out;

    // workspace layout
    float* stateA   = (float*)d_ws;                         // N
    float* stateB   = stateA + N_NODES;                     // N
    float* partials = stateB + N_NODES;                     // T * 128
    u32*   cnt      = (u32*)(partials + (size_t)T_STEPS * TOTAL_WAVES);

    // allow >64KB dynamic LDS (no-op if already set; not a stream op)
    (void)hipFuncSetAttribute((const void*)esn_persistent,
                              hipFuncAttributeMaxDynamicSharedMemorySize, LDS_BYTES);

    esn_init<<<(N_NODES + 255) / 256, 256, 0, stream>>>(state0, stateA, cnt);

    esn_persistent<<<NUM_WG, THREADS_PER_WG, LDS_BYTES, stream>>>(
        X, Y, W_in, W_res, W_fb, W_out, stateA, stateB, partials, cnt);

    esn_readout<<<(T_STEPS + 255) / 256, 256, 0, stream>>>(X, W_out, b_out,
                                                           partials, out);
}
